// Encoder_68453188763927
// MI455X (gfx1250) — compile-verified, matrix operand order checked by reading, not run on hardware
//
#include <hip/hip_runtime.h>
#include <hip/hip_bf16.h>
#include <stdint.h>

// ---------------------------------------------------------------------------
// MI455X / gfx1250 implementation of the GNN encoder.
// All 128-wide dense linears run through v_wmma_f32_16x16x32_bf16 with
// weights pre-packed into WMMA B-fragment layout (loaded as b128 pairs).
// GEMM blocks cover 32 rows: each wave keeps two f32 accumulators sharing a
// single B-fragment load (2x WMMA density, half the weight traffic).
// The per-edge conv message (bond embed + GN + grouped gate/val + post GEMM
// + segment-sum) is fused into one kernel scattering through L2 f32 atomics;
// the entire working set fits in the 192MB L2, so this is compute-bound.
// ---------------------------------------------------------------------------

#define WDIM 128
#define NGRAPH 1024
#define NLAYER 3
#define KSTEP  4
#define CHOP   2
#define MAXDEG 4
#define ATOMV  128
#define BONDV  8
#define NFA    9
#define NFB    3

typedef __attribute__((ext_vector_type(16))) __bf16 v16bf;
typedef __attribute__((ext_vector_type(8)))  float  v8f;

__device__ __forceinline__ __bf16 f2bf(float f) {
  unsigned u = __builtin_bit_cast(unsigned, f);
  unsigned r = (u + 0x7FFFu + ((u >> 16) & 1u)) >> 16;
  unsigned short s = (unsigned short)r;
  return __builtin_bit_cast(__bf16, s);
}

// ---------------------------------------------------------------------------
// Weight packing: f32 [OUT][KIN] -> bf16 fragments.
// Fragment (nt,kt) is a 16(N) x 32(K) tile; lane l keeps 16 bf16 in the
// CDNA5 16-bit operand layout: n = nt*16 + (l&15), half h = l>>4,
// VGPR v -> K = kt*32 + (v&4 ? 16:0) + 8*h + 2*(v&3) (+pair).
// Stored 512 bf16/fragment: frag*512 + lane*16 + 2v(+p)  (=> 2x b128/lane).
// ---------------------------------------------------------------------------
__global__ void pack_weight_kernel(const float* __restrict__ w,
                                   __bf16* __restrict__ p, int OUT, int KIN) {
  int idx = blockIdx.x * blockDim.x + threadIdx.x;
  int KT = KIN >> 5;
  int frags = (OUT >> 4) * KT;
  if (idx >= frags * 32) return;
  int frag = idx >> 5, lane = idx & 31;
  int nt = frag / KT, kt = frag - nt * KT;
  int n = nt * 16 + (lane & 15);
  int h = lane >> 4;
  __bf16* dst = p + (size_t)frag * 512 + lane * 16;
#pragma unroll
  for (int v = 0; v < 8; ++v) {
    int kk = kt * 32 + ((v & 4) ? 16 : 0) + 8 * h + 2 * (v & 3);
    dst[2 * v]     = f2bf(w[(size_t)n * KIN + kk]);
    dst[2 * v + 1] = f2bf(w[(size_t)n * KIN + kk + 1]);
  }
}

// Dual row-tile WMMA accumulate: C0 += A0*B, C1 += A1*B, B fragment loaded
// once per k-tile. Fully unrolled K loop (KIN compile-time).
template <int KIN>
__device__ __forceinline__ void wmma_rowtile2(const __bf16* ldsA0,
                                              const __bf16* ldsA1,
                                              const __bf16* __restrict__ Wp,
                                              int nt, v8f& c0, v8f& c1) {
  int lane = threadIdx.x & 31;
  int m = lane & 15, h = lane >> 4;
  constexpr int KT = KIN >> 5;
#pragma unroll
  for (int kt = 0; kt < KT; ++kt) {
    v16bf b = *(const v16bf*)(Wp + ((size_t)(nt * KT + kt) * 512) + lane * 16);
    if (kt + 1 < KT)  // pull next fragment toward the WGP (global_prefetch_b8)
      __builtin_prefetch(Wp + ((size_t)(nt * KT + kt + 1) * 512) + lane * 16, 0, 3);
    const __bf16* ar0 = ldsA0 + m * KIN + kt * 32 + 8 * h;
    const __bf16* ar1 = ldsA1 + m * KIN + kt * 32 + 8 * h;
    v16bf a0, a1;
#pragma unroll
    for (int v = 0; v < 8; ++v) {
      int kk = ((v & 4) ? 16 : 0) + 2 * (v & 3);
      a0[2 * v]     = ar0[kk];
      a0[2 * v + 1] = ar0[kk + 1];
      a1[2 * v]     = ar1[kk];
      a1[2 * v + 1] = ar1[kk + 1];
    }
    c0 = __builtin_amdgcn_wmma_f32_16x16x32_bf16(false, a0, false, b,
                                                 (short)0, c0, false, false);
    c1 = __builtin_amdgcn_wmma_f32_16x16x32_bf16(false, a1, false, b,
                                                 (short)0, c1, false, false);
  }
}

// ---------------------------------------------------------------------------
// Generic Y[M,128] = X[M,KIN] @ W^T (+bias) (*exp(scale)) (+=)
// block = 256 threads = 8 waves; 32 rows/block; wave w owns columns 16w..16w+15
// with two accumulators (rows 0-15 and 16-31 of the block tile).
// ---------------------------------------------------------------------------
template <int KIN>
__global__ void linear_wmma_kernel(const float* __restrict__ X,
                                   const __bf16* __restrict__ Wp,
                                   const float* __restrict__ bias,
                                   const float* __restrict__ scalevec,
                                   float* __restrict__ Y,
                                   int M, int accmode) {
  __shared__ __bf16 ldsA[32 * KIN];
  int row0 = blockIdx.x * 32;
  for (int i = threadIdx.x; i < 32 * KIN; i += 256) {
    int r = i / KIN, ccol = i - r * KIN;
    float x = (row0 + r < M) ? X[(size_t)(row0 + r) * KIN + ccol] : 0.f;
    ldsA[r * KIN + ccol] = f2bf(x);
  }
  __syncthreads();
  int wave = threadIdx.x >> 5;
  v8f c0 = {}, c1 = {};
  wmma_rowtile2<KIN>(ldsA, ldsA + 16 * KIN, Wp, wave, c0, c1);
  int lane = threadIdx.x & 31;
  int n = wave * 16 + (lane & 15);
  int h = lane >> 4;
  float bb = bias ? bias[n] : 0.f;
  float sc = scalevec ? __expf(scalevec[n]) : 1.f;
#pragma unroll
  for (int r = 0; r < 8; ++r) {
    int row = row0 + r + 8 * h;
    if (row < M) {
      float v = (c0[r] + bb) * sc;
      size_t o = (size_t)row * WDIM + n;
      if (accmode) Y[o] += v; else Y[o] = v;
    }
    int row1 = row0 + 16 + r + 8 * h;
    if (row1 < M) {
      float v = (c1[r] + bb) * sc;
      size_t o = (size_t)row1 * WDIM + n;
      if (accmode) Y[o] += v; else Y[o] = v;
    }
  }
}

// ---------------------------------------------------------------------------
// Fused conv edge-message kernel: 32 edges / block (256 thr = 8 waves).
// bond-embed + (xp[dst]+b, xp[src]+b) -> GN(16 groups of 8) -> grouped 8x8
// gate/val -> relu*mul -> WMMA 128x128 post GEMM (dual tile) -> atomic
// segment-sum into xnew.
// ---------------------------------------------------------------------------
__global__ void conv_edge_kernel(const float* __restrict__ xp,
                                 const int* __restrict__ ei,   // [2,E]
                                 const int* __restrict__ ea,   // [E,3]
                                 const float* __restrict__ bemb, // [3][8][128]
                                 const float* __restrict__ gw,   // [16][8][8]
                                 const float* __restrict__ vw,   // [16][8][8]
                                 const __bf16* __restrict__ postWp,
                                 const float* __restrict__ postb,
                                 float* __restrict__ xnew, int E) {
  __shared__ float xi[32][WDIM];
  __shared__ float xj[32][WDIM];
  __shared__ __bf16 A[32 * WDIM];
  int t = threadIdx.x;
  int e0 = blockIdx.x * 32;
  { // load + bond embed: thread t -> edge t>>3, 16 channels at (t&7)*16
    int el = t >> 3, cb = (t & 7) * 16;
    int ge = e0 + el;
    if (ge < E) {
      int a0 = ea[ge * 3 + 0], a1 = ea[ge * 3 + 1], a2 = ea[ge * 3 + 2];
      int s = ei[ge], d = ei[E + ge];
      const float* b0 = bemb + (size_t)a0 * WDIM;
      const float* b1 = bemb + (size_t)(BONDV + a1) * WDIM;
      const float* b2 = bemb + (size_t)(2 * BONDV + a2) * WDIM;
      const float* xs = xp + (size_t)s * WDIM;
      const float* xd = xp + (size_t)d * WDIM;
#pragma unroll
      for (int j = 0; j < 16; ++j) {
        int cc = cb + j;
        float b = b0[cc] + b1[cc] + b2[cc];
        xi[el][cc] = xd[cc] + b;
        xj[el][cc] = xs[cc] + b;
      }
    } else {
#pragma unroll
      for (int j = 0; j < 16; ++j) { xi[el][cb + j] = 0.f; xj[el][cb + j] = 0.f; }
    }
  }
  __syncthreads();
  { // GroupNorm (16 groups of 8) on xi and xj: 512 tasks, 2 per thread
#pragma unroll
    for (int tk = 0; tk < 2; ++tk) {
      int task = t * 2 + tk;
      int el = task >> 4, g8 = (task & 15) * 8;
      for (int which = 0; which < 2; ++which) {
        float* p = which ? &xj[el][g8] : &xi[el][g8];
        float m = 0.f;
#pragma unroll
        for (int i2 = 0; i2 < 8; ++i2) m += p[i2];
        m *= 0.125f;
        float v = 0.f;
#pragma unroll
        for (int i2 = 0; i2 < 8; ++i2) { float d = p[i2] - m; v += d * d; }
        v *= 0.125f;
        float r = rsqrtf(v + 1e-5f);
#pragma unroll
        for (int i2 = 0; i2 < 8; ++i2) p[i2] = (p[i2] - m) * r;
      }
    }
  }
  __syncthreads();
  { // grouped conv: gate(xi) relu * val(xj); 512 group-tasks, 2 per thread
#pragma unroll
    for (int tk = 0; tk < 2; ++tk) {
      int task = t * 2 + tk;
      int el = task >> 4, g = task & 15;
      const float* gi = &xi[el][g * 8];
      const float* vi = &xj[el][g * 8];
#pragma unroll
      for (int o = 0; o < 8; ++o) {
        const float* gr = gw + (g * 8 + o) * 8;
        const float* vr = vw + (g * 8 + o) * 8;
        float ga = 0.f, va = 0.f;
#pragma unroll
        for (int i2 = 0; i2 < 8; ++i2) { ga += gr[i2] * gi[i2]; va += vr[i2] * vi[i2]; }
        ga = ga > 0.f ? ga : 0.f;
        A[el * WDIM + g * 8 + o] = f2bf(ga * va);
      }
    }
  }
  __syncthreads();
  int wave = t >> 5;
  v8f c0 = {}, c1 = {};
  wmma_rowtile2<WDIM>(A, A + 16 * WDIM, postWp, wave, c0, c1);
  int lane = t & 31;
  int n = wave * 16 + (lane & 15);
  int h = lane >> 4;
  float bb = postb[n];
#pragma unroll
  for (int r = 0; r < 8; ++r) {
    int em = e0 + r + 8 * h;
    if (em < E)
      atomicAdd(&xnew[(size_t)ei[E + em] * WDIM + n], c0[r] + bb);
    int em1 = e0 + 16 + r + 8 * h;
    if (em1 < E)
      atomicAdd(&xnew[(size_t)ei[E + em1] * WDIM + n], c1[r] + bb);
  }
}

// ---------------------------------------------------------------------------
// Small VALU / glue kernels
// ---------------------------------------------------------------------------
__global__ void zero_f32_kernel(float* p, long n) {
  long i = (long)blockIdx.x * blockDim.x + threadIdx.x;
  if (i < n) p[i] = 0.f;
}
__global__ void zero_i32_kernel(int* p, long n) {
  long i = (long)blockIdx.x * blockDim.x + threadIdx.x;
  if (i < n) p[i] = 0;
}
__global__ void copy_f32_kernel(float* d, const float* s, long n) {
  long i = (long)blockIdx.x * blockDim.x + threadIdx.x;
  if (i < n) d[i] = s[i];
}
__global__ void add2_kernel(float* d, const float* a, const float* b, long n) {
  long i = (long)blockIdx.x * blockDim.x + threadIdx.x;
  if (i < n) d[i] = a[i] + b[i];
}
__global__ void hop_reset_kernel(float* xraw, float* xout, const float* hin, long n) {
  long i = (long)blockIdx.x * blockDim.x + threadIdx.x;
  if (i < n) { xraw[i] = hin[i] + xout[i]; xout[i] = 0.f; }
}
__global__ void conv_scale_update_kernel(float* xout, const float* xnew,
                                         const float* scale /*[4][W]*/,
                                         const int* deg, long n) {
  long i = (long)blockIdx.x * blockDim.x + threadIdx.x;
  if (i < n) {
    int node = (int)(i >> 7), cc = (int)(i & 127);
    xout[i] += __expf(scale[deg[node] * WDIM + cc]) * xnew[i];
  }
}
__global__ void groupnorm_kernel(float* X, int M, int C, int G) {
  int idx = blockIdx.x * blockDim.x + threadIdx.x;
  if (idx >= M * G) return;
  int row = idx / G, g = idx - row * G;
  int n = C / G;
  float* p = X + (size_t)row * C + g * n;
  float m = 0.f;
  for (int i = 0; i < n; ++i) m += p[i];
  m /= n;
  float v = 0.f;
  for (int i = 0; i < n; ++i) { float d = p[i] - m; v += d * d; }
  v /= n;
  float r = rsqrtf(v + 1e-5f);
  for (int i = 0; i < n; ++i) p[i] = (p[i] - m) * r;
}
__global__ void gconv_gate_val_kernel(const float* __restrict__ X,
                                      const float* __restrict__ gw,
                                      const float* __restrict__ vw,
                                      float* __restrict__ Y,
                                      int M, int G, int OPG, int IPG) {
  int OUTC = G * OPG;
  long idx = (long)blockIdx.x * blockDim.x + threadIdx.x;
  if (idx >= (long)M * OUTC) return;
  int row = (int)(idx / OUTC), o = (int)(idx - (long)row * OUTC);
  int g = o / OPG, oi = o - g * OPG;
  const float* xr = X + (size_t)row * (G * IPG) + g * IPG;
  const float* gr = gw + (size_t)(g * OPG + oi) * IPG;
  const float* vr = vw + (size_t)(g * OPG + oi) * IPG;
  float a = 0.f, b = 0.f;
  for (int i = 0; i < IPG; ++i) { a += gr[i] * xr[i]; b += vr[i] * xr[i]; }
  a = a > 0.f ? a : 0.f;
  Y[(size_t)row * OUTC + o] = a * b;
}
__global__ void segsum_rows_kernel(float* dst, const float* src,
                                   const int* batch, long n) {
  long i = (long)blockIdx.x * blockDim.x + threadIdx.x;
  if (i < n) {
    int node = (int)(i >> 7), cc = (int)(i & 127);
    atomicAdd(&dst[(size_t)batch[node] * WDIM + cc], src[i]);
  }
}
__global__ void gather_add_rows_kernel(float* dst, const float* src,
                                       const int* batch, long n) {
  long i = (long)blockIdx.x * blockDim.x + threadIdx.x;
  if (i < n) {
    int node = (int)(i >> 7), cc = (int)(i & 127);
    dst[i] += src[(size_t)batch[node] * WDIM + cc];
  }
}
__global__ void deg_count_kernel(int* deg, const int* ei, int E) {
  int e = blockIdx.x * blockDim.x + threadIdx.x;
  if (e < E) atomicAdd(&deg[ei[E + e]], 1);
}
__global__ void deg_clip_kernel(int* deg, int N) {
  int i = blockIdx.x * blockDim.x + threadIdx.x;
  if (i < N) { int d = deg[i] - 1; d = d < 0 ? 0 : d; deg[i] = d > 3 ? 3 : d; }
}
__global__ void bond0_scatter_kernel(float* h0, const int* ei, const int* ea,
                                     const float* be, long n, int E) {
  long i = (long)blockIdx.x * blockDim.x + threadIdx.x;
  if (i < n) {
    int e = (int)(i >> 7), cc = (int)(i & 127);
    int d = ei[E + e];
    float b = be[(size_t)ea[e * 3 + 0] * WDIM + cc] +
              be[(size_t)(BONDV + ea[e * 3 + 1]) * WDIM + cc] +
              be[(size_t)(2 * BONDV + ea[e * 3 + 2]) * WDIM + cc];
    atomicAdd(&h0[(size_t)d * WDIM + cc], b);
  }
}
__global__ void node_embed_kernel(float* hin, const int* x, const int* deg,
                                  const float* aemb, const float* scale0,
                                  const float* h0, long n) {
  long i = (long)blockIdx.x * blockDim.x + threadIdx.x;
  if (i < n) {
    int node = (int)(i >> 7), cc = (int)(i & 127);
    float s = 0.f;
#pragma unroll
    for (int f = 0; f < NFA; ++f)
      s += aemb[((size_t)f * ATOMV + x[node * NFA + f]) * WDIM + cc];
    s += __expf(scale0[deg[node] * WDIM + cc]) * h0[i];
    hin[i] = s;
  }
}

// ---------------------------------------------------------------------------
// Linear attention: one block per graph (batch sorted -> binary search for
// node range). k,v computed cooperatively per node; kv outer products and
// ksum accumulated in registers/LDS with zero atomics. hatt carries kv state
// across layers (kv = segsum + h_att; h_att = kv).
// ---------------------------------------------------------------------------
__global__ void att_kv_kernel(const float* __restrict__ xv,
                              const int* __restrict__ batch,
                              const float* __restrict__ kw,
                              const float* __restrict__ vw,
                              float* __restrict__ hatt,
                              float* __restrict__ ksum, int N) {
  int b = blockIdx.x, t = threadIdx.x;
  int lo = 0, hi = N;
  while (lo < hi) { int mid = (lo + hi) >> 1; if (batch[mid] < b) lo = mid + 1; else hi = mid; }
  int s = lo;
  lo = s; hi = N;
  while (lo < hi) { int mid = (lo + hi) >> 1; if (batch[mid] < b + 1) lo = mid + 1; else hi = mid; }
  int e = lo;
  __shared__ float kls[256], vls[256];
  float acc[16];
#pragma unroll
  for (int i = 0; i < 16; ++i) acc[i] = 0.f;
  float ks = 0.f;
  const float inv = 0.25f; // 1/sqrt(HD=16)
  int g = t >> 5, oi = t & 31;
  const float* kr = kw + (size_t)(g * 32 + oi) * 16;
  const float* vr = vw + (size_t)(g * 32 + oi) * 16;
  int hh = t >> 4, aa = t & 15;
  for (int i = s; i < e; ++i) {
    const float* xg = xv + (size_t)i * WDIM + g * 16;
    float ka = 0.f, va = 0.f;
#pragma unroll
    for (int j = 0; j < 16; ++j) { ka += kr[j] * xg[j]; va += vr[j] * xg[j]; }
    __syncthreads();
    kls[t] = __expf(ka * inv);
    vls[t] = va;
    __syncthreads();
    ks += kls[t];
    float kval = kls[hh * 16 + aa];
#pragma unroll
    for (int j = 0; j < 16; ++j) acc[j] += kval * vls[hh * 16 + j];
  }
  float* kvg = hatt + (size_t)b * 4096 + t * 16;
#pragma unroll
  for (int j = 0; j < 16; ++j) kvg[j] += acc[j];
  ksum[(size_t)b * 256 + t] = ks;
}

__global__ void att_out_kernel(const float* __restrict__ xv,
                               const int* __restrict__ batch,
                               const float* __restrict__ qw,
                               const float* __restrict__ hatt,
                               const float* __restrict__ ksum,
                               float* __restrict__ out, int N) {
  __shared__ float qls[16][256];
  __shared__ float den[16][16];
  int t = threadIdx.x;
  int n0 = blockIdx.x * 16;
  const float inv = 0.25f;
  { // phase 1: q = exp(gconv(xv)/sqrt(hd))
    int nd = t >> 4, cb = (t & 15) * 16;
    int gn = n0 + nd;
    const float* x = (gn < N) ? xv + (size_t)gn * WDIM : xv;
    for (int j = 0; j < 16; ++j) {
      int o = cb + j, g = o >> 5, oi = o & 31;
      const float* qr = qw + (size_t)(g * 32 + oi) * 16;
      const float* xg = x + g * 16;
      float a = 0.f;
#pragma unroll
      for (int jj = 0; jj < 16; ++jj) a += qr[jj] * xg[jj];
      qls[nd][o] = __expf(a * inv);
    }
  }
  __syncthreads();
  { // phase 2: denominators q . ksum per (node, head)
    int nd = t >> 4, hh = t & 15;
    int gn = n0 + nd;
    int bi = (gn < N) ? batch[gn] : 0;
    const float* kss = ksum + (size_t)bi * 256 + hh * 16;
    float d = 0.f;
#pragma unroll
    for (int j = 0; j < 16; ++j) d += qls[nd][hh * 16 + j] * kss[j];
    den[nd][hh] = d;
  }
  __syncthreads();
  { // phase 3: out[node, hh*16+vv] = (q . kv)/den
    int nd = t >> 4, vv = t & 15;
    int gn = n0 + nd;
    if (gn < N) {
      int bi = batch[gn];
      const float* kvb = hatt + (size_t)bi * 4096;
      float* og = out + (size_t)gn * 256;
      for (int hh = 0; hh < 16; ++hh) {
        const float* kvh = kvb + hh * 256 + vv;
        const float* qh = &qls[nd][hh * 16];
        float a = 0.f;
#pragma unroll
        for (int h = 0; h < 16; ++h) a += qh[h] * kvh[h * 16];
        og[hh * 16 + vv] = a / den[nd][hh];
      }
    }
  }
}

// ---------------------------------------------------------------------------
// Host orchestration
// ---------------------------------------------------------------------------
static inline dim3 gsz(long n, int b) { return dim3((unsigned)((n + b - 1) / b)); }

extern "C" void kernel_launch(void* const* d_in, const int* in_sizes, int n_in,
                              void* d_out, int out_size, void* d_ws, size_t ws_size,
                              hipStream_t stream) {
  const int* xin   = (const int*)d_in[0];
  const int* ei    = (const int*)d_in[1];
  const int* ea    = (const int*)d_in[2];
  const int* batch = (const int*)d_in[3];
  const float* atom_emb   = (const float*)d_in[4];
  const float* bond_emb0  = (const float*)d_in[5];
  const float* scale0     = (const float*)d_in[6];
  const float* conv_bemb  = (const float*)d_in[7];
  const float* conv_pre_w = (const float*)d_in[8];
  const float* conv_gw    = (const float*)d_in[9];
  const float* conv_vw    = (const float*)d_in[10];
  const float* conv_post_w= (const float*)d_in[11];
  const float* conv_post_b= (const float*)d_in[12];
  const float* conv_scale = (const float*)d_in[13];
  const float* virt_pre_w = (const float*)d_in[14];
  const float* virt_pre_b = (const float*)d_in[15];
  const float* virt_gw    = (const float*)d_in[16];
  const float* virt_vw    = (const float*)d_in[17];
  const float* virt_post_w= (const float*)d_in[18];
  const float* virt_post_b= (const float*)d_in[19];
  const float* virt_scale = (const float*)d_in[20];
  const float* att_pre_w  = (const float*)d_in[21];
  const float* att_pre_b  = (const float*)d_in[22];
  const float* att_qw     = (const float*)d_in[23];
  const float* att_kw     = (const float*)d_in[24];
  const float* att_vw     = (const float*)d_in[25];
  const float* att_post_w = (const float*)d_in[26];
  const float* att_post_b = (const float*)d_in[27];
  const float* att_scale  = (const float*)d_in[28];
  const float* main_pre_w = (const float*)d_in[29];
  const float* main_pre_b = (const float*)d_in[30];
  const float* main_gw    = (const float*)d_in[31];
  const float* main_vw    = (const float*)d_in[32];
  const float* main_post_w= (const float*)d_in[33];
  const float* main_post_b= (const float*)d_in[34];

  const int N = in_sizes[0] / NFA;
  const int E = in_sizes[1] / 2;
  const long NW = (long)N * WDIM;

  // ---- workspace carve-out --------------------------------------------
  char* base = (char*)d_ws;
  size_t off = 0;
  auto allocB = [&](size_t bytes) -> char* {
    char* p = base + off;
    off += (bytes + 255) & ~(size_t)255;
    return p;
  };
  float* hin  = (float*)allocB(NW * 4);
  float* hout = (float*)allocB(NW * 4);
  float* xraw = (float*)allocB(NW * 4);
  float* xout = (float*)allocB(NW * 4);
  float* xp   = (float*)allocB(NW * 4);
  float* xnew = (float*)allocB(NW * 4);
  float* tmp1 = (float*)allocB(NW * 4);
  float* tmp3 = (float*)allocB(NW * 3 * 4);
  float* bxx  = (float*)allocB((size_t)NGRAPH * WDIM * 4);
  float* bpre = (float*)allocB((size_t)NGRAPH * WDIM * 4);
  float* bgv  = (float*)allocB((size_t)NGRAPH * 2 * WDIM * 4);
  float* bout = (float*)allocB((size_t)NGRAPH * WDIM * 4);
  float* hvirt= (float*)allocB((size_t)NGRAPH * WDIM * 4);
  float* hatt = (float*)allocB((size_t)NGRAPH * 4096 * 4);
  float* ksum = (float*)allocB((size_t)NGRAPH * 256 * 4);
  int*   deg  = (int*)allocB((size_t)N * 4);
  __bf16* p_conv_pre  = (__bf16*)allocB((size_t)NLAYER * KSTEP * 128 * 128 * 2);
  __bf16* p_conv_post = (__bf16*)allocB((size_t)NLAYER * KSTEP * 128 * 128 * 2);
  __bf16* p_virt_pre  = (__bf16*)allocB((size_t)NLAYER * 128 * 128 * 2);
  __bf16* p_virt_post = (__bf16*)allocB((size_t)NLAYER * 128 * 256 * 2);
  __bf16* p_att_pre   = (__bf16*)allocB((size_t)NLAYER * 128 * 128 * 2);
  __bf16* p_att_post  = (__bf16*)allocB((size_t)NLAYER * 128 * 256 * 2);
  __bf16* p_main_pre  = (__bf16*)allocB((size_t)NLAYER * 128 * 128 * 2);
  __bf16* p_main_post = (__bf16*)allocB((size_t)NLAYER * 128 * 384 * 2);
  if (off > ws_size) return; // workspace too small: bail deterministically

  auto pack = [&](const float* w, __bf16* p, int OUT, int KIN) {
    int th = (OUT / 16) * (KIN / 32) * 32;
    pack_weight_kernel<<<gsz(th, 256), 256, 0, stream>>>(w, p, OUT, KIN);
  };
  auto lin = [&](const float* X, const __bf16* Wp, const float* bias,
                 const float* sv, float* Y, int M, int KIN, int acc) {
    dim3 g((unsigned)(((long)M + 31) / 32));
    if (KIN == 128)
      linear_wmma_kernel<128><<<g, 256, 0, stream>>>(X, Wp, bias, sv, Y, M, acc);
    else if (KIN == 256)
      linear_wmma_kernel<256><<<g, 256, 0, stream>>>(X, Wp, bias, sv, Y, M, acc);
    else
      linear_wmma_kernel<384><<<g, 256, 0, stream>>>(X, Wp, bias, sv, Y, M, acc);
  };

  // ---- 1. pack all dense weights to bf16 WMMA fragments ---------------
  for (int l = 0; l < NLAYER; ++l) {
    for (int k = 0; k < KSTEP; ++k) {
      size_t o = (size_t)(l * KSTEP + k) * 128 * 128;
      pack(conv_pre_w + o,  p_conv_pre + o,  128, 128);
      pack(conv_post_w + o, p_conv_post + o, 128, 128);
    }
    pack(virt_pre_w + (size_t)l * 128 * 128,  p_virt_pre + (size_t)l * 128 * 128,  128, 128);
    pack(virt_post_w + (size_t)l * 128 * 256, p_virt_post + (size_t)l * 128 * 256, 128, 256);
    pack(att_pre_w + (size_t)l * 128 * 128,   p_att_pre + (size_t)l * 128 * 128,   128, 128);
    pack(att_post_w + (size_t)l * 128 * 256,  p_att_post + (size_t)l * 128 * 256,  128, 256);
    pack(main_pre_w + (size_t)l * 128 * 128,  p_main_pre + (size_t)l * 128 * 128,  128, 128);
    pack(main_post_w + (size_t)l * 128 * 384, p_main_post + (size_t)l * 128 * 384, 128, 384);
  }

  // ---- 2. init: degree, h0 scatter, node embedding --------------------
  zero_i32_kernel<<<gsz(N, 256), 256, 0, stream>>>(deg, N);
  zero_f32_kernel<<<gsz(NW, 256), 256, 0, stream>>>(xnew, NW); // xnew = h0
  deg_count_kernel<<<gsz(E, 256), 256, 0, stream>>>(deg, ei, E);
  bond0_scatter_kernel<<<gsz((long)E * WDIM, 256), 256, 0, stream>>>(
      xnew, ei, ea, bond_emb0, (long)E * WDIM, E);
  deg_clip_kernel<<<gsz(N, 256), 256, 0, stream>>>(deg, N);
  node_embed_kernel<<<gsz(NW, 256), 256, 0, stream>>>(
      hin, xin, deg, atom_emb, scale0, xnew, NW);
  zero_f32_kernel<<<gsz((long)NGRAPH * WDIM, 256), 256, 0, stream>>>(
      hvirt, (long)NGRAPH * WDIM);
  zero_f32_kernel<<<gsz((long)NGRAPH * 4096, 256), 256, 0, stream>>>(
      hatt, (long)NGRAPH * 4096);

  // ---- 3. layers ------------------------------------------------------
  float* praw = xraw;  // conv x_raw
  float* pnew = xnew;  // conv segment-sum target
  for (int l = 0; l < NLAYER; ++l) {
    // ---------- conv_msg ----------
    copy_f32_kernel<<<gsz(NW, 256), 256, 0, stream>>>(praw, hin, NW);
    zero_f32_kernel<<<gsz(NW, 256), 256, 0, stream>>>(xout, NW);
    for (int k = 0; k < KSTEP; ++k) {
      if (k > 0 && (k % CHOP) == 0)
        hop_reset_kernel<<<gsz(NW, 256), 256, 0, stream>>>(praw, xout, hin, NW);
      size_t wo = (size_t)(l * KSTEP + k) * 128 * 128;
      lin(praw, p_conv_pre + wo, nullptr, nullptr, xp, N, 128, 0);
      zero_f32_kernel<<<gsz(NW, 256), 256, 0, stream>>>(pnew, NW);
      conv_edge_kernel<<<gsz(((long)E + 31) / 32, 1).x, 256, 0, stream>>>(
          xp, ei, ea,
          conv_bemb + (size_t)(l * KSTEP + k) * NFB * BONDV * WDIM,
          conv_gw + (size_t)(l * KSTEP + k) * 16 * 8 * 8,
          conv_vw + (size_t)(l * KSTEP + k) * 16 * 8 * 8,
          p_conv_post + wo,
          conv_post_b + (size_t)(l * KSTEP + k) * WDIM,
          pnew, E);
      conv_scale_update_kernel<<<gsz(NW, 256), 256, 0, stream>>>(
          xout, pnew, conv_scale + (size_t)(l * KSTEP + k) * MAXDEG * WDIM, deg, NW);
      float* t = praw; praw = pnew; pnew = t; // x_raw = segment_sum result
    }
    add2_kernel<<<gsz(NW, 256), 256, 0, stream>>>(hout, hin, xout, NW);

    // ---------- virt_msg ----------
    copy_f32_kernel<<<gsz((long)NGRAPH * WDIM, 256), 256, 0, stream>>>(
        bxx, hvirt, (long)NGRAPH * WDIM);
    segsum_rows_kernel<<<gsz(NW, 256), 256, 0, stream>>>(bxx, hin, batch, NW);
    copy_f32_kernel<<<gsz((long)NGRAPH * WDIM, 256), 256, 0, stream>>>(
        hvirt, bxx, (long)NGRAPH * WDIM);
    lin(bxx, p_virt_pre + (size_t)l * 128 * 128,
        virt_pre_b + (size_t)l * WDIM, nullptr, bpre, NGRAPH, 128, 0);
    groupnorm_kernel<<<gsz((long)NGRAPH * 16, 256), 256, 0, stream>>>(
        bpre, NGRAPH, 128, 16);
    gconv_gate_val_kernel<<<gsz((long)NGRAPH * 256, 256), 256, 0, stream>>>(
        bpre, virt_gw + (size_t)l * 16 * 16 * 8, virt_vw + (size_t)l * 16 * 16 * 8,
        bgv, NGRAPH, 16, 16, 8);
    lin(bgv, p_virt_post + (size_t)l * 128 * 256,
        virt_post_b + (size_t)l * WDIM, virt_scale + (size_t)l * WDIM,
        bout, NGRAPH, 256, 0);
    gather_add_rows_kernel<<<gsz(NW, 256), 256, 0, stream>>>(hout, bout, batch, NW);

    // ---------- att_msg ----------
    lin(hin, p_att_pre + (size_t)l * 128 * 128,
        att_pre_b + (size_t)l * WDIM, nullptr, tmp1, N, 128, 0);
    groupnorm_kernel<<<gsz((long)N * 8, 256), 256, 0, stream>>>(tmp1, N, 128, 8);
    att_kv_kernel<<<NGRAPH, 256, 0, stream>>>(
        tmp1, batch, att_kw + (size_t)l * 8 * 32 * 16,
        att_vw + (size_t)l * 8 * 32 * 16, hatt, ksum, N);
    att_out_kernel<<<gsz(((long)N + 15) / 16, 1).x, 256, 0, stream>>>(
        tmp1, batch, att_qw + (size_t)l * 8 * 32 * 16, hatt, ksum, tmp3, N);
    lin(tmp3, p_att_post + (size_t)l * 128 * 256,
        att_post_b + (size_t)l * WDIM, att_scale + (size_t)l * WDIM,
        hout, N, 256, 1); // hout += exp(scale)*(out@W^T+b)

    // ---------- main GLB ----------
    lin(hout, p_main_pre + (size_t)l * 128 * 128,
        main_pre_b + (size_t)l * WDIM, nullptr, tmp1, N, 128, 0);
    groupnorm_kernel<<<gsz((long)N * 16, 256), 256, 0, stream>>>(tmp1, N, 128, 16);
    gconv_gate_val_kernel<<<gsz((long)N * 384, 256), 256, 0, stream>>>(
        tmp1, main_gw + (size_t)l * 16 * 24 * 8, main_vw + (size_t)l * 16 * 24 * 8,
        tmp3, N, 16, 24, 8);
    lin(tmp3, p_main_post + (size_t)l * 128 * 384,
        main_post_b + (size_t)l * WDIM, nullptr, hin, N, 384, 0);
  }

  // ---- 4. output ------------------------------------------------------
  copy_f32_kernel<<<gsz(NW, 256), 256, 0, stream>>>((float*)d_out, hin, NW);
}
